// GeneEmbedding_6150393168706
// MI455X (gfx1250) — matrix operand order repaired and checked by reading, not compiled, not run on hardware
//
#include <hip/hip_runtime.h>

typedef __attribute__((ext_vector_type(16))) _Float16 v16h;
typedef __attribute__((ext_vector_type(8)))  _Float16 v8h;
typedef __attribute__((ext_vector_type(8)))  float    v8f;
typedef __attribute__((ext_vector_type(4)))  float    v4f;

#define GDIM   1024
#define DDIM   32
#define NBATCH 128          // B*L
#define LN_EPS 1e-3f

#define WMMA(A, B, C) \
  __builtin_amdgcn_wmma_f32_16x16x32_f16(false, (A), false, (B), (short)0, (C), false, false)

// 16-bit A-matrix 16x32 layout used here: lane holds row (lane&15);
// element e -> K = e + (lane>=16 ? 16 : 0)
// => per lane, 16 contiguous elements starting at K = hsel*16.

// Build A-frag from an f32 row (32 floats): 4x b128 LDS loads + cvt.
__device__ __forceinline__ v16h load_afrag_f32(const float* rowp, int hsel) {
  const v4f* p = (const v4f*)(rowp + hsel * 16);
  v4f a0 = p[0], a1 = p[1];        // K = off .. off+7
  v4f a2 = p[2], a3 = p[3];        // K = off+8 .. off+15
  v16h r;
#pragma unroll
  for (int j = 0; j < 4; ++j) {
    r[j]      = (_Float16)a0[j];
    r[4 + j]  = (_Float16)a1[j];
    r[8 + j]  = (_Float16)a2[j];
    r[12 + j] = (_Float16)a3[j];
  }
  return r;
}

// Build A-frag from a staged f16 row (32 halves): 2x b128 LDS loads.
__device__ __forceinline__ v16h load_afrag_f16(const _Float16* rowp, int hsel) {
  const v8h* q = (const v8h*)(rowp + hsel * 16);
  v8h lo = q[0];                   // K = off .. off+7
  v8h hi = q[1];                   // K = off+8 .. off+15
  return __builtin_shufflevector(lo, hi, 0, 1, 2, 3, 4, 5, 6, 7,
                                 8, 9, 10, 11, 12, 13, 14, 15);
}

// B-frag (32x16): lane holds column (lane&15), elements K = e + hsel*16,
// i.e. 16 contiguous halves: 2x b128 LDS loads.
__device__ __forceinline__ v16h load_bfrag(const _Float16* p) {
  const v8h* q = (const v8h*)p;
  v8h lo = q[0], hi = q[1];
  return __builtin_shufflevector(lo, hi, 0, 1, 2, 3, 4, 5, 6, 7,
                                 8, 9, 10, 11, 12, 13, 14, 15);
}

// reduce across the 16 lanes of a half-wave (wave32)
__device__ __forceinline__ float hred_max(float x) {
  x = fmaxf(x, __shfl_xor(x, 1, 32));
  x = fmaxf(x, __shfl_xor(x, 2, 32));
  x = fmaxf(x, __shfl_xor(x, 4, 32));
  x = fmaxf(x, __shfl_xor(x, 8, 32));
  return x;
}
__device__ __forceinline__ float hred_sum(float x) {
  x += __shfl_xor(x, 1, 32);
  x += __shfl_xor(x, 2, 32);
  x += __shfl_xor(x, 4, 32);
  x += __shfl_xor(x, 8, 32);
  return x;
}

// ---------------------------------------------------------------------------
// Stage 1: gating pipeline. One thread per (b,l,g) row.
// ---------------------------------------------------------------------------
__global__ void __launch_bounds__(256)
prep_kernel(const float* __restrict__ x,
            const float* __restrict__ pkt,    // (G, D)
            const float* __restrict__ pv,     // (D,)
            const float* __restrict__ pk2,    // (D, D)
            const float* __restrict__ disc,   // (D,)
            const float* __restrict__ sfp,    // scalar
            float* __restrict__ out_token,    // (N*G, D)
            float* __restrict__ hbuf)         // (N*G, D)
{
  __shared__ float pk2s[DDIM * DDIM];
  __shared__ float pvs[DDIM], discs[DDIM];
  const int tid = threadIdx.x;
  for (int i = tid; i < DDIM * DDIM; i += 256) pk2s[i] = pk2[i];
  if (tid < DDIM) { pvs[tid] = pv[tid]; discs[tid] = disc[tid]; }
  __syncthreads();

  const float sf = sfp[0];
  const int row = blockIdx.x * 256 + tid;           // 0 .. N*G-1 (exact grid)
  const int g = row & (GDIM - 1);
  const float xv = x[row];

  float val[DDIM], tmp[DDIM];
#pragma unroll
  for (int d = 0; d < DDIM; ++d) {
    float t = xv * pvs[d];
    val[d] = (t >= 0.f) ? t : 0.1f * t;             // leaky_relu(0.1)
  }
  for (int d = 0; d < DDIM; ++d) {
    float acc = sf * val[d];
#pragma unroll
    for (int e = 0; e < DDIM; ++e) acc += val[e] * pk2s[e * DDIM + d];
    tmp[d] = acc;
  }
  float mx = -1e30f;
#pragma unroll
  for (int d = 0; d < DDIM; ++d) mx = fmaxf(mx, tmp[d]);
  float s = 0.f;
#pragma unroll
  for (int d = 0; d < DDIM; ++d) { tmp[d] = __expf(tmp[d] - mx); s += tmp[d]; }
  const float inv = 1.f / s;
  float scal = 0.f;
#pragma unroll
  for (int d = 0; d < DDIM; ++d) scal += tmp[d] * inv * discs[d];

  const float* tk = pkt + (size_t)g * DDIM;
  float* ot = out_token + (size_t)row * DDIM;
  float* oh = hbuf + (size_t)row * DDIM;
#pragma unroll
  for (int d = 0; d < DDIM; ++d) {
    float tkd = tk[d];
    ot[d] = tkd;
    oh[d] = tkd * scal;
  }
}

// ---------------------------------------------------------------------------
// Stage 2: fused flash-attention block, in place on hbuf.
// One workgroup (256 threads = 8 waves) per batch item.
// LDS: hs (f32 1024x32, 128KB) | K row-major (f16, 64KB) | V^T (f16, 64KB)
//      | per-wave 16x32 f16 staging (8KB)  => 270336 bytes dynamic LDS.
// ---------------------------------------------------------------------------
__global__ void __launch_bounds__(256)
attn_kernel(float* __restrict__ hbuf,
            const float* __restrict__ Wq, const float* __restrict__ bq,
            const float* __restrict__ Wk, const float* __restrict__ bk,
            const float* __restrict__ Wv, const float* __restrict__ bv,
            const float* __restrict__ Wo, const float* __restrict__ bo,
            const float* __restrict__ gam, const float* __restrict__ bet)
{
  extern __shared__ char smem[];
  float*    hs  = (float*)smem;                                   // 1024*32 f32
  _Float16* Ks  = (_Float16*)(smem + GDIM * DDIM * 4);            // [g][d]
  _Float16* Vt  = (_Float16*)(smem + GDIM * DDIM * 6);            // [d][g]
  _Float16* stg = (_Float16*)(smem + GDIM * DDIM * 8);            // 8 * 16*32

  const int bn   = blockIdx.x;
  const int tid  = threadIdx.x;
  const int lane = tid & 31;
  const int wave = tid >> 5;
  const int col  = lane & 15;
  const int hsel = (lane >= 16) ? 1 : 0;

  __builtin_prefetch(Wq, 0, 3);
  __builtin_prefetch(Wk, 0, 3);
  __builtin_prefetch(Wv, 0, 3);
  __builtin_prefetch(Wo, 0, 3);

  float* hg = hbuf + (size_t)bn * GDIM * DDIM;

  // Async copy of the 128KB h slice straight into LDS (no VGPR round-trip).
  // Per-lane LDS byte address goes in the "VDST" VGPR (low 32 bits of the
  // generic shared pointer are the LDS byte offset); global addr is a VGPR
  // pair; tracked by ASYNCcnt.
  {
    const unsigned ldsbase = (unsigned)(size_t)hs;
#pragma unroll 4
    for (int it = 0; it < (GDIM * DDIM) / (256 * 4); ++it) {
      const int i = (it * 256 + tid) * 4;           // 4 floats = 16B per lane
      const unsigned lofs = ldsbase + (unsigned)i * 4u;
      const float* gp = hg + i;
      asm volatile("global_load_async_to_lds_b128 %0, %1, off"
                   :: "v"(lofs), "v"(gp) : "memory");
    }
    asm volatile("s_wait_asynccnt 0x0" ::: "memory");
  }
  __syncthreads();

  _Float16* mystg = stg + wave * (16 * DDIM);
  v8f z = {};

  // ---- Phase 1: K = h@Wk + bk (row-major f16), V^T = (h@Wv + bv)^T ----
  v16h wkb0, wkb1, wvb0, wvb1;
#pragma unroll
  for (int e = 0; e < 16; ++e) {
    int kk = e + hsel * 16;                       // B-frag: B[kk][col]
    wkb0[e] = (_Float16)Wk[kk * DDIM + col];
    wkb1[e] = (_Float16)Wk[kk * DDIM + col + 16];
    wvb0[e] = (_Float16)Wv[kk * DDIM + col];
    wvb1[e] = (_Float16)Wv[kk * DDIM + col + 16];
  }
  const float bk0 = bk[col], bk1 = bk[col + 16];
  const float bv0 = bv[col], bv1 = bv[col + 16];

  for (int t = wave; t < GDIM / 16; t += 8) {
    const int r0 = t * 16;
    v16h a = load_afrag_f32(&hs[(r0 + col) * DDIM], hsel);
    v8f k0 = WMMA(a, wkb0, z);
    v8f k1 = WMMA(a, wkb1, z);
    v8f v0 = WMMA(a, wvb0, z);
    v8f v1 = WMMA(a, wvb1, z);
#pragma unroll
    for (int v = 0; v < 8; ++v) {                 // K rows: scatter once
      int m = r0 + v + hsel * 8;
      Ks[m * DDIM + col]      = (_Float16)(k0[v] + bk0);
      Ks[m * DDIM + col + 16] = (_Float16)(k1[v] + bk1);
    }
    v8h vp0, vp1;                                 // V^T: contiguous b128 store
#pragma unroll
    for (int v = 0; v < 8; ++v) {
      vp0[v] = (_Float16)(v0[v] + bv0);
      vp1[v] = (_Float16)(v1[v] + bv1);
    }
    *(v8h*)&Vt[col * GDIM + r0 + hsel * 8]        = vp0;
    *(v8h*)&Vt[(col + 16) * GDIM + r0 + hsel * 8] = vp1;
  }
  __syncthreads();

  // ---- Phase 2: per query tile: Q proj, flash attention, O proj, LN ----
  v16h wqb0, wqb1, wob0, wob1;
#pragma unroll
  for (int e = 0; e < 16; ++e) {
    int kk = e + hsel * 16;
    wqb0[e] = (_Float16)Wq[kk * DDIM + col];
    wqb1[e] = (_Float16)Wq[kk * DDIM + col + 16];
    wob0[e] = (_Float16)Wo[kk * DDIM + col];
    wob1[e] = (_Float16)Wo[kk * DDIM + col + 16];
  }
  const float bq0 = bq[col], bq1 = bq[col + 16];
  const float bo0 = bo[col], bo1 = bo[col + 16];
  const float g0 = gam[col], g1 = gam[col + 16];
  const float be0 = bet[col], be1 = bet[col + 16];
  const float qscale = 0.17677669529663687f;      // 1/sqrt(32)

  for (int qt = wave; qt < GDIM / 16; qt += 8) {
    const int q0 = qt * 16;

    // Q tile (scaled), staged C->A via per-wave LDS (same-wave DS in order)
    v16h a = load_afrag_f32(&hs[(q0 + col) * DDIM], hsel);
    v8f qc0 = WMMA(a, wqb0, z);
    v8f qc1 = WMMA(a, wqb1, z);
#pragma unroll
    for (int v = 0; v < 8; ++v) {
      int m = v + hsel * 8;
      mystg[m * DDIM + col]      = (_Float16)((qc0[v] + bq0) * qscale);
      mystg[m * DDIM + col + 16] = (_Float16)((qc1[v] + bq1) * qscale);
    }
    v16h qa = load_afrag_f16(&mystg[col * DDIM], hsel);

    v8f o0 = {}, o1 = {};
    float mrun[8], lrun[8];
#pragma unroll
    for (int v = 0; v < 8; ++v) { mrun[v] = -1e30f; lrun[v] = 0.f; }

    for (int c = 0; c < GDIM / 32; ++c) {          // 32 keys per chunk
      const int kb = c * 32;
      v16h kbf0 = load_bfrag(&Ks[(kb + col) * DDIM + hsel * 16]);
      v16h kbf1 = load_bfrag(&Ks[(kb + 16 + col) * DDIM + hsel * 16]);
      v8f s0 = WMMA(qa, kbf0, z);
      v8f s1 = WMMA(qa, kbf1, z);

      float p0[8], p1[8];
#pragma unroll
      for (int v = 0; v < 8; ++v) {
        float mx = hred_max(fmaxf(s0[v], s1[v]));
        float mnew = fmaxf(mrun[v], mx);
        float fsc = __expf(mrun[v] - mnew);
        p0[v] = __expf(s0[v] - mnew);
        p1[v] = __expf(s1[v] - mnew);
        float rs = hred_sum(p0[v] + p1[v]);
        lrun[v] = lrun[v] * fsc + rs;
        mrun[v] = mnew;
        o0[v] *= fsc;
        o1[v] *= fsc;
      }
      // stage P (16 q-rows x 32 keys) C->A
#pragma unroll
      for (int v = 0; v < 8; ++v) {
        int m = v + hsel * 8;
        mystg[m * DDIM + col]      = (_Float16)p0[v];
        mystg[m * DDIM + col + 16] = (_Float16)p1[v];
      }
      v16h pa = load_afrag_f16(&mystg[col * DDIM], hsel);
      v16h vbf0 = load_bfrag(&Vt[col * GDIM + kb + hsel * 16]);
      v16h vbf1 = load_bfrag(&Vt[(col + 16) * GDIM + kb + hsel * 16]);
      o0 = WMMA(pa, vbf0, o0);                     // accumulate into C
      o1 = WMMA(pa, vbf1, o1);
    }

    // normalize, stage O C->A, project with Wo
#pragma unroll
    for (int v = 0; v < 8; ++v) {
      float inv = 1.f / lrun[v];
      int m = v + hsel * 8;
      mystg[m * DDIM + col]      = (_Float16)(o0[v] * inv);
      mystg[m * DDIM + col + 16] = (_Float16)(o1[v] * inv);
    }
    v16h oa = load_afrag_f16(&mystg[col * DDIM], hsel);
    v8f y0 = WMMA(oa, wob0, z);
    v8f y1 = WMMA(oa, wob1, z);

    // bias + residual + layernorm, write back (LDS hs keeps pristine input)
#pragma unroll
    for (int v = 0; v < 8; ++v) {
      int m = q0 + v + hsel * 8;
      float a0 = y0[v] + bo0 + hs[m * DDIM + col];
      float a1 = y1[v] + bo1 + hs[m * DDIM + col + 16];
      float mean = hred_sum(a0 + a1) * (1.f / 32.f);
      float d0 = a0 - mean, d1 = a1 - mean;
      float var = hred_sum(d0 * d0 + d1 * d1) * (1.f / 32.f);
      float rs = rsqrtf(var + LN_EPS);
      hg[m * DDIM + col]      = g0 * d0 * rs + be0;
      hg[m * DDIM + col + 16] = g1 * d1 * rs + be1;
    }
  }
}

extern "C" void kernel_launch(void* const* d_in, const int* in_sizes, int n_in,
                              void* d_out, int out_size, void* d_ws, size_t ws_size,
                              hipStream_t stream) {
  (void)in_sizes; (void)n_in; (void)out_size; (void)d_ws; (void)ws_size;
  const float* x    = (const float*)d_in[0];
  const float* pkt  = (const float*)d_in[1];
  const float* pv   = (const float*)d_in[2];
  const float* pk2  = (const float*)d_in[3];
  const float* disc = (const float*)d_in[4];
  const float* sf   = (const float*)d_in[5];
  const float* Wq = (const float*)d_in[6];  const float* bq = (const float*)d_in[7];
  const float* Wk = (const float*)d_in[8];  const float* bk = (const float*)d_in[9];
  const float* Wv = (const float*)d_in[10]; const float* bv = (const float*)d_in[11];
  const float* Wo = (const float*)d_in[12]; const float* bo = (const float*)d_in[13];
  const float* gam = (const float*)d_in[14];
  const float* bet = (const float*)d_in[15];

  float* out_token = (float*)d_out;                         // (B,L,G,D)
  float* hbuf = out_token + (size_t)NBATCH * GDIM * DDIM;   // second output half

  prep_kernel<<<NBATCH * GDIM / 256, 256, 0, stream>>>(
      x, pkt, pv, pk2, disc, sf, out_token, hbuf);

  const size_t smem = (size_t)GDIM * DDIM * 4   // hs  f32
                    + (size_t)GDIM * DDIM * 2   // K   f16
                    + (size_t)GDIM * DDIM * 2   // V^T f16
                    + 8 * 16 * DDIM * 2;        // staging
  attn_kernel<<<NBATCH, 256, smem, stream>>>(hbuf, Wq, bq, Wk, bk, Wv, bv,
                                             Wo, bo, gam, bet);
  attn_kernel<<<NBATCH, 256, smem, stream>>>(hbuf, Wq, bq, Wk, bk, Wv, bv,
                                             Wo, bo, gam, bet);
}
